// SerializedMultiHeadMultiLayerAttention_37830071943575
// MI455X (gfx1250) — matrix operand order repaired and checked by reading, not compiled, not run on hardware
//
#include <hip/hip_runtime.h>
#include <hip/hip_bf16.h>
#include <math.h>

// Problem constants (from reference setup_inputs)
#define B_  64
#define D_  256
#define L_  2048
#define NB_ 2
#define H_  512
#define BL_ (B_ * L_)

typedef __bf16 bf16_t;
typedef __attribute__((ext_vector_type(16))) __bf16 v16bf;
typedef __attribute__((ext_vector_type(8)))  float  v8f;

// ---------------------------------------------------------------------------
// CDNA5 async global->LDS copy (ASYNCcnt-tracked DMA, bypasses VGPRs).
// Builtin signature takes <4 x i32> pointers in AS(1)/AS(3).
// Falls back to a synchronous load+ds_store if the builtin is not declared.
// ---------------------------------------------------------------------------
#if defined(__has_builtin)
#if __has_builtin(__builtin_amdgcn_global_load_async_to_lds_b128) && \
    __has_builtin(__builtin_amdgcn_s_wait_asynccnt)
#define USE_ASYNC_LDS 1
#endif
#endif

#if defined(USE_ASYNC_LDS)
typedef int v4i_gcc __attribute__((vector_size(16)));
typedef __attribute__((address_space(1))) v4i_gcc as1_v4i;
typedef __attribute__((address_space(3))) v4i_gcc as3_v4i;
#define ASYNC_WAIT() __builtin_amdgcn_s_wait_asynccnt(0)
#else
#define ASYNC_WAIT()
#endif

__device__ __forceinline__ void cp16_to_lds(const bf16_t* g, bf16_t* l) {
#if defined(USE_ASYNC_LDS)
    __builtin_amdgcn_global_load_async_to_lds_b128((as1_v4i*)g, (as3_v4i*)l, 0, 0);
#else
    *(uint4*)l = *(const uint4*)g;
#endif
}

// ---------------------------------------------------------------------------
// small utility kernels
// ---------------------------------------------------------------------------
__global__ __launch_bounds__(256) void zero_kernel(float* p, int n) {
    int i = blockIdx.x * 256 + threadIdx.x;
    if (i < n) p[i] = 0.f;
}

__global__ __launch_bounds__(256) void cvt_kernel(const float* __restrict__ in,
                                                  bf16_t* __restrict__ out, int n) {
    int i = blockIdx.x * 256 + threadIdx.x;
    if (i < n) out[i] = (bf16_t)in[i];
}

// ---------------------------------------------------------------------------
// LN over D of v^T with LDS tile transpose.
//   src    : (B, D, L) f32    [optionally + corr[b,d] broadcast over l]
//   out    : (B*L, D)  bf16 row-major
//   colsum : optional per-(b,d) sum / sumsq accumulation over L (atomics)
// One workgroup handles a (32 l) x (256 d) tile.
// ---------------------------------------------------------------------------
__global__ __launch_bounds__(256)
void ln_transpose_kernel(const float* __restrict__ src, const float* __restrict__ corr,
                         const float* __restrict__ gamma, const float* __restrict__ beta,
                         bf16_t* __restrict__ out,
                         float* __restrict__ colsum, float* __restrict__ colsum2)
{
    __shared__ float tile[32][257];           // padded: stride 257 -> conflict free
    __shared__ float red1[32][8], red2[32][8];
    __shared__ float mu_s[32], rs_s[32];

    const int t  = threadIdx.x;
    const int b  = blockIdx.x / (L_ / 32);
    const int l0 = (blockIdx.x % (L_ / 32)) * 32;
    const int ll = t & 31;
    const int wv = t >> 5;

    // load tile coalesced along L (lanes = consecutive l)
    for (int it = 0; it < 32; ++it) {
        int d = it * 8 + wv;
        float c = corr ? corr[b * D_ + d] : 0.f;
        tile[ll][d] = src[(size_t)b * D_ * L_ + (size_t)d * L_ + l0 + ll] + c;
    }
    __syncthreads();

    // per-l mean/var over D (8 partial segments per l)
    {
        int l = t >> 3, seg = t & 7;
        float s = 0.f, s2 = 0.f;
        #pragma unroll 8
        for (int j = 0; j < 32; ++j) {
            float x = tile[l][seg * 32 + j];
            s += x; s2 += x * x;
        }
        red1[l][seg] = s; red2[l][seg] = s2;
    }
    __syncthreads();
    if (t < 32) {
        float s = 0.f, s2 = 0.f;
        #pragma unroll
        for (int j = 0; j < 8; ++j) { s += red1[t][j]; s2 += red2[t][j]; }
        float mu  = s * (1.f / D_);
        float var = s2 * (1.f / D_) - mu * mu;
        mu_s[t] = mu;
        rs_s[t] = rsqrtf(var + 1e-6f);
    }
    __syncthreads();

    // write bf16 rows coalesced along D; accumulate column stats over the 32 l's
    const int d = t;
    const float g = gamma[d], bb = beta[d];
    float cs = 0.f, cs2 = 0.f;
    for (int l = 0; l < 32; ++l) {
        float x = (tile[l][d] - mu_s[l]) * rs_s[l] * g + bb;
        out[((size_t)b * L_ + l0 + l) * D_ + d] = (bf16_t)x;
        cs += x; cs2 += x * x;
    }
    if (colsum) {
        atomicAdd(&colsum[b * D_ + d], cs);
        atomicAdd(&colsum2[b * D_ + d], cs2);
    }
}

// ---------------------------------------------------------------------------
// finalize per-(b,d) mu/sd over L and compute q = [mu;sd] @ wq^T  (B x 128)
// ---------------------------------------------------------------------------
__global__ __launch_bounds__(128)
void stats_q_kernel(const float* __restrict__ colsum, const float* __restrict__ colsum2,
                    const float* __restrict__ wq, float* __restrict__ qout)
{
    __shared__ float mu_s[D_], sd_s[D_];
    const int b = blockIdx.x, t = threadIdx.x;
    for (int d = t; d < D_; d += 128) {
        float mu  = colsum[b * D_ + d] * (1.f / L_);
        float var = colsum2[b * D_ + d] * (1.f / L_) - mu * mu;
        mu_s[d] = mu;
        sd_s[d] = sqrtf(fmaxf(var, 1e-10f));
    }
    __syncthreads();
    float acc = 0.f;
    const float* wr = wq + (size_t)t * (2 * D_);
    for (int d = 0; d < D_; ++d)
        acc += mu_s[d] * wr[d] + sd_s[d] * wr[D_ + d];
    qout[b * 128 + t] = acc;
}

// ---------------------------------------------------------------------------
// bf16 WMMA GEMM:  C(M,N) = A(M,K) * Bw(N,K)^T  [+bias] [relu]
// Workgroup: 256 threads (8 waves), tile 128(M) x 64(N), K-step 64,
// double-buffered LDS staged with async global->LDS DMA (ASYNCcnt).
// Wave grid 4x2; each wave computes a 32x32 block = 2x2 WMMA 16x16 tiles.
// Fragment packing uses contiguous K (k = 16*half + j) in A(16x32) and B(32x16) alike.
// ---------------------------------------------------------------------------
#define KSTEP 64
#define ALD   (KSTEP + 8)      // padded LDS row stride (bf16 elems)

template <bool RELU, bool OUTBF>
__global__ __launch_bounds__(256)
void gemm_bf16(const bf16_t* __restrict__ A, const bf16_t* __restrict__ Bw,
               const float* __restrict__ bias, void* __restrict__ Cout,
               int N, int K)
{
    __shared__ __align__(16) bf16_t As[2][128 * ALD];   // 2 x (128 rows x 64 K)
    __shared__ __align__(16) bf16_t Bs[2][64 * ALD];    // 2 x (64 rows  x 64 K)

    const int t    = threadIdx.x;
    const int lane = t & 31;
    const int wave = t >> 5;
    const int wm   = wave & 3;          // 4 waves along M (32 rows each)
    const int wn   = wave >> 2;         // 2 waves along N (32 cols each)
    const int m0   = blockIdx.x * 128;
    const int n0   = blockIdx.y * 64;

    const int half = lane >> 4;         // K-half selector (contiguous K: 16*half + j)
    const int mr   = lane & 15;

    v8f acc[2][2];
    for (int i = 0; i < 2; ++i)
        for (int j = 0; j < 2; ++j)
            for (int r = 0; r < 8; ++r) acc[i][j][r] = 0.f;

    // staging: A tile 128x64 -> 32 elems/thread (4x b128); B tile 64x64 -> 16 elems (2x b128)
    const int arow = t >> 1, acol = (t & 1) * 32;
    const int brow = t >> 2, bcol = (t & 3) * 16;

    auto stage = [&](int k0, int buf) {
        const bf16_t* ag = A + (size_t)(m0 + arow) * K + k0 + acol;
        bf16_t* al = &As[buf][arow * ALD + acol];
        cp16_to_lds(ag,      al);
        cp16_to_lds(ag + 8,  al + 8);
        cp16_to_lds(ag + 16, al + 16);
        cp16_to_lds(ag + 24, al + 24);
        const bf16_t* bg = Bw + (size_t)(n0 + brow) * K + k0 + bcol;
        bf16_t* bl = &Bs[buf][brow * ALD + bcol];
        cp16_to_lds(bg,     bl);
        cp16_to_lds(bg + 8, bl + 8);
    };

    stage(0, 0);
    ASYNC_WAIT();
    __syncthreads();

    int buf = 0;
    for (int k0 = 0; k0 < K; k0 += KSTEP) {
        if (k0 + KSTEP < K) stage(k0 + KSTEP, buf ^ 1);   // DMA next tile (overlaps WMMA)

        #pragma unroll
        for (int kk = 0; kk < KSTEP; kk += 32) {
            v16bf afrag[2], bfrag[2];
            #pragma unroll
            for (int ms = 0; ms < 2; ++ms) {
                // lane<16: M=lane, K=0..15; lane>=16: M=lane-16, K=16..31
                const bf16_t* p = &As[buf][(wm * 32 + ms * 16 + mr) * ALD + kk + half * 16];
                #pragma unroll
                for (int j = 0; j < 16; ++j) afrag[ms][j] = p[j];
            }
            #pragma unroll
            for (int ns = 0; ns < 2; ++ns) {
                // lane<16: N=lane, K=0..15; lane>=16: N=lane-16, K=16..31
                const bf16_t* p = &Bs[buf][(wn * 32 + ns * 16 + mr) * ALD + kk + half * 16];
                #pragma unroll
                for (int j = 0; j < 16; ++j) bfrag[ns][j] = p[j];
            }
            #pragma unroll
            for (int ms = 0; ms < 2; ++ms)
                #pragma unroll
                for (int ns = 0; ns < 2; ++ns)
                    acc[ms][ns] = __builtin_amdgcn_wmma_f32_16x16x32_bf16(
                        false, afrag[ms], false, bfrag[ns],
                        (short)0, acc[ms][ns], false, false);
        }

        ASYNC_WAIT();          // own async transfers for next tile complete
        __syncthreads();       // cross-wave: safe to read next buf / overwrite this buf
        buf ^= 1;
    }

    // C layout: VGPR r -> M = r + 8*(lane>=16), N = lane&15
    const int nn = lane & 15;
    #pragma unroll
    for (int ms = 0; ms < 2; ++ms)
        #pragma unroll
        for (int ns = 0; ns < 2; ++ns) {
            const int col = n0 + wn * 32 + ns * 16 + nn;
            const float bv = bias ? bias[col] : 0.f;
            #pragma unroll
            for (int r = 0; r < 8; ++r) {
                const int row = m0 + wm * 32 + ms * 16 + half * 8 + r;
                float v = acc[ms][ns][r] + bv;
                if (RELU) v = fmaxf(v, 0.f);
                const size_t idx = (size_t)row * N + col;
                if (OUTBF) ((bf16_t*)Cout)[idx] = (bf16_t)v;
                else       ((float*)Cout)[idx]  = v;
            }
        }
}

// ---------------------------------------------------------------------------
// scores = k.q * scale -> softmax over L -> weighted mean/std per (b,d)
// one workgroup per batch b
// ---------------------------------------------------------------------------
__global__ __launch_bounds__(256)
void attn_stats_kernel(const float* __restrict__ kproj, const float* __restrict__ qv,
                       const bf16_t* __restrict__ vtbf,
                       float* __restrict__ att_mean, float* __restrict__ att_std)
{
    __shared__ float sc[L_];
    __shared__ float qs[128];
    __shared__ float red[256];
    const int b = blockIdx.x, t = threadIdx.x;
    const int lane = t & 31, wave = t >> 5;
    const float scale = 0.088388347648318447f;  // 1/sqrt(128)

    if (t < 128) qs[t] = qv[b * 128 + t];
    __syncthreads();

    // wave-parallel dot over 128-d k rows (float4 per lane + xor reduce)
    for (int l = wave; l < L_; l += 8) {
        const float4 kv = *(const float4*)&kproj[((size_t)b * L_ + l) * 128 + lane * 4];
        float p = kv.x * qs[lane * 4 + 0] + kv.y * qs[lane * 4 + 1] +
                  kv.z * qs[lane * 4 + 2] + kv.w * qs[lane * 4 + 3];
        #pragma unroll
        for (int off = 16; off > 0; off >>= 1) p += __shfl_xor(p, off, 32);
        if (lane == 0) sc[l] = p * scale;
    }
    __syncthreads();

    // softmax over L
    float m = -1e30f;
    for (int l = t; l < L_; l += 256) m = fmaxf(m, sc[l]);
    red[t] = m; __syncthreads();
    for (int s = 128; s > 0; s >>= 1) { if (t < s) red[t] = fmaxf(red[t], red[t + s]); __syncthreads(); }
    m = red[0];
    __syncthreads();
    float ssum = 0.f;
    for (int l = t; l < L_; l += 256) { float e = __expf(sc[l] - m); sc[l] = e; ssum += e; }
    red[t] = ssum; __syncthreads();
    for (int s = 128; s > 0; s >>= 1) { if (t < s) red[t] += red[t + s]; __syncthreads(); }
    const float inv = 1.f / red[0];
    __syncthreads();
    for (int l = t; l < L_; l += 256) sc[l] *= inv;
    __syncthreads();

    // weighted stats per d (coalesced: consecutive threads = consecutive d)
    const int d = t;
    float wsum = 0.f, wsum2 = 0.f;
    for (int l = 0; l < L_; ++l) {
        const float w = sc[l];
        const float x = (float)vtbf[((size_t)b * L_ + l) * D_ + d];
        wsum += w * x; wsum2 += w * x * x;
    }
    const float mean = wsum;
    const float var  = wsum2 - mean * mean;   // sum w == 1
    att_mean[b * D_ + d] = mean;
    att_std[b * D_ + d]  = sqrtf(fmaxf(var, 1e-10f));
}

// ---------------------------------------------------------------------------
// skip_sum += fc_w @ [mean;std] + fc_b ; corr = fcq_w @ mean
// ---------------------------------------------------------------------------
__global__ __launch_bounds__(256)
void skip_corr_kernel(const float* __restrict__ att_mean, const float* __restrict__ att_std,
                      const float* __restrict__ fc_w, const float* __restrict__ fc_b,
                      const float* __restrict__ fcq_w,
                      float* __restrict__ skip_sum, float* __restrict__ corr)
{
    __shared__ float m_s[D_], s_s[D_];
    const int b = blockIdx.x, o = threadIdx.x;
    m_s[o] = att_mean[b * D_ + o];
    s_s[o] = att_std[b * D_ + o];
    __syncthreads();
    float accS = fc_b[o], accC = 0.f;
    const float* fr = fc_w  + (size_t)o * (2 * D_);
    const float* qr = fcq_w + (size_t)o * D_;
    for (int d = 0; d < D_; ++d) {
        accS += fr[d] * m_s[d] + fr[D_ + d] * s_s[d];
        accC += qr[d] * m_s[d];
    }
    skip_sum[b * D_ + o] += accS;   // unique (b,o) per thread
    corr[b * D_ + o] = accC;
}

// ---------------------------------------------------------------------------
// v_new(B,D,L) = h2(B*L,D)^T + v_src + corr  (LDS tile transpose; in-place ok)
// ---------------------------------------------------------------------------
__global__ __launch_bounds__(256)
void residual_kernel(const float* __restrict__ h2, const float* __restrict__ vsrc,
                     const float* __restrict__ corr, float* __restrict__ vdst)
{
    __shared__ float tile[32][257];
    const int t  = threadIdx.x;
    const int b  = blockIdx.x / (L_ / 32);
    const int l0 = (blockIdx.x % (L_ / 32)) * 32;
    for (int r = 0; r < 32; ++r)
        tile[r][t] = h2[((size_t)b * L_ + l0 + r) * D_ + t];
    __syncthreads();
    const int ll = t & 31, wv = t >> 5;
    for (int it = 0; it < 32; ++it) {
        const int d = it * 8 + wv;
        const size_t idx = (size_t)b * D_ * L_ + (size_t)d * L_ + l0 + ll;
        vdst[idx] = tile[ll][d] + vsrc[idx] + corr[b * D_ + d];
    }
}

// ---------------------------------------------------------------------------
// final: relu + BatchNorm over batch axis per channel d
// ---------------------------------------------------------------------------
__global__ __launch_bounds__(256)
void final_bn_kernel(const float* __restrict__ skip_sum, float* __restrict__ out)
{
    const int d = threadIdx.x;
    float s = 0.f, s2 = 0.f;
    for (int b = 0; b < B_; ++b) {
        float y = fmaxf(skip_sum[b * D_ + d], 0.f);
        s += y; s2 += y * y;
    }
    const float m   = s * (1.f / B_);
    const float var = s2 * (1.f / B_) - m * m;
    const float rs  = rsqrtf(var + 1e-5f);
    for (int b = 0; b < B_; ++b) {
        float y = fmaxf(skip_sum[b * D_ + d], 0.f);
        out[b * D_ + d] = (y - m) * rs;
    }
}

// ---------------------------------------------------------------------------
extern "C" void kernel_launch(void* const* d_in, const int* in_sizes, int n_in,
                              void* d_out, int out_size, void* d_ws, size_t ws_size,
                              hipStream_t stream)
{
    (void)in_sizes; (void)n_in; (void)out_size; (void)ws_size;
    const float* x     = (const float*)d_in[0];
    const float* wk    = (const float*)d_in[1];
    const float* wq    = (const float*)d_in[2];
    const float* fcq_w = (const float*)d_in[3];
    const float* fc_w  = (const float*)d_in[4];
    const float* fc_b  = (const float*)d_in[5];
    const float* att_g = (const float*)d_in[6];
    const float* att_b = (const float*)d_in[7];
    const float* w1    = (const float*)d_in[8];
    const float* b1    = (const float*)d_in[9];
    const float* w2    = (const float*)d_in[10];
    const float* b2    = (const float*)d_in[11];
    const float* ffw_g = (const float*)d_in[12];
    const float* ffw_b = (const float*)d_in[13];

    char* ws = (char*)d_ws;
    size_t off = 0;
    auto alloc = [&](size_t bytes) -> char* {
        char* p = ws + off;
        off += (bytes + 255) & ~(size_t)255;
        return p;
    };
    float*  v      = (float*)alloc((size_t)B_ * D_ * L_ * 4);     // evolving state
    bf16_t* actbf  = (bf16_t*)alloc((size_t)BL_ * D_ * 2);        // vt / ffn-in (reused)
    bf16_t* h1bf   = (bf16_t*)alloc((size_t)BL_ * H_ * 2);
    float*  h2     = (float*)alloc((size_t)BL_ * D_ * 4);
    float*  kproj  = (float*)alloc((size_t)BL_ * 128 * 4);
    bf16_t* wkbf   = (bf16_t*)alloc((size_t)128 * D_ * 2);
    bf16_t* w1bf   = (bf16_t*)alloc((size_t)H_ * D_ * 2);
    bf16_t* w2bf   = (bf16_t*)alloc((size_t)D_ * H_ * 2);
    float*  colsum = (float*)alloc((size_t)2 * B_ * D_ * 4);
    float*  colsum2 = colsum + B_ * D_;
    float*  qv     = (float*)alloc((size_t)B_ * 128 * 4);
    float*  attm   = (float*)alloc((size_t)B_ * D_ * 4);
    float*  atts   = (float*)alloc((size_t)B_ * D_ * 4);
    float*  corr   = (float*)alloc((size_t)B_ * D_ * 4);
    float*  skip   = (float*)alloc((size_t)B_ * D_ * 4);

    zero_kernel<<<(B_ * D_ + 255) / 256, 256, 0, stream>>>(skip, B_ * D_);

    const float* vsrc = x;
    for (int i = 0; i < NB_; ++i) {
        zero_kernel<<<(2 * B_ * D_ + 255) / 256, 256, 0, stream>>>(colsum, 2 * B_ * D_);
        cvt_kernel<<<(128 * D_ + 255) / 256, 256, 0, stream>>>(wk + (size_t)i * 128 * D_, wkbf, 128 * D_);
        cvt_kernel<<<(H_ * D_ + 255) / 256, 256, 0, stream>>>(w1 + (size_t)i * H_ * D_, w1bf, H_ * D_);
        cvt_kernel<<<(D_ * H_ + 255) / 256, 256, 0, stream>>>(w2 + (size_t)i * D_ * H_, w2bf, D_ * H_);

        // attention-side LN (+ column stats for mu/sd over L)
        ln_transpose_kernel<<<B_ * (L_ / 32), 256, 0, stream>>>(
            vsrc, nullptr, att_g + i * D_, att_b + i * D_, actbf, colsum, colsum2);
        stats_q_kernel<<<B_, 128, 0, stream>>>(colsum, colsum2,
            wq + (size_t)i * 128 * 2 * D_, qv);
        // k = vt @ wk^T   (WMMA)
        gemm_bf16<false, false><<<dim3(BL_ / 128, 128 / 64), 256, 0, stream>>>(
            actbf, wkbf, nullptr, kproj, 128, D_);
        attn_stats_kernel<<<B_, 256, 0, stream>>>(kproj, qv, actbf, attm, atts);
        skip_corr_kernel<<<B_, 256, 0, stream>>>(attm, atts,
            fc_w + (size_t)i * D_ * 2 * D_, fc_b + i * D_,
            fcq_w + (size_t)i * D_ * D_, skip, corr);

        // FFN: LN(q_out^T) -> relu(.@w1^T+b1) -> .@w2^T+b2 -> residual transpose-add
        ln_transpose_kernel<<<B_ * (L_ / 32), 256, 0, stream>>>(
            vsrc, corr, ffw_g + i * D_, ffw_b + i * D_, actbf, nullptr, nullptr);
        gemm_bf16<true, true><<<dim3(BL_ / 128, H_ / 64), 256, 0, stream>>>(
            actbf, w1bf, b1 + i * H_, h1bf, H_, D_);
        gemm_bf16<false, false><<<dim3(BL_ / 128, D_ / 64), 256, 0, stream>>>(
            h1bf, w2bf, b2 + i * D_, h2, D_, H_);
        residual_kernel<<<B_ * (L_ / 32), 256, 0, stream>>>(h2, vsrc, corr, v);
        vsrc = v;
    }
    final_bn_kernel<<<1, 256, 0, stream>>>(skip, (float*)d_out);
}
